// C2QAttention_15281493639176
// MI455X (gfx1250) — matrix operand order repaired and checked by reading, not compiled, not run on hardware
//
#include <hip/hip_runtime.h>

// Problem constants (from reference): B=32, C=4096, Q=512, D=512
#define B_ 32
#define C_ 4096
#define Q_ 512
#define D_ 512

typedef __attribute__((ext_vector_type(16))) __bf16 v16bf;
typedef __attribute__((ext_vector_type(8)))  __bf16 v8bf;
typedef __attribute__((ext_vector_type(4)))  __bf16 v4bf;
typedef __attribute__((ext_vector_type(8)))  float  v8f;
typedef __attribute__((ext_vector_type(4)))  float  v4f;

union BU { v16bf v; v8bf h[2]; };

// ---------------------------------------------------------------------------
// Pre-pass: transpose + convert encoded_question [B,Q,D] f32 -> [B,D,Q] bf16
// so WMMA B-fragments are contiguous loads. 32x32 LDS tiles. Only ~50 MB r/w.
// ---------------------------------------------------------------------------
__global__ __launch_bounds__(256) void transpose_to_bf16(
    const float* __restrict__ E, __bf16* __restrict__ ET) {
  __shared__ __bf16 tile[32][33];   // +1 pad avoids bank conflicts
  const int b  = blockIdx.z;
  const int d0 = blockIdx.x * 32;
  const int q0 = blockIdx.y * 32;
  const int tx = threadIdx.x & 31;
  const int ty = threadIdx.x >> 5;  // 0..7

  const float* __restrict__ src = E + ((long long)b * Q_ + q0) * D_ + d0;
#pragma unroll
  for (int r = 0; r < 32; r += 8)
    tile[ty + r][tx] = (__bf16)src[(long long)(ty + r) * D_ + tx];
  __syncthreads();

  __bf16* __restrict__ dst = ET + ((long long)b * D_ + d0) * Q_ + q0;
#pragma unroll
  for (int r = 0; r < 32; r += 8)
    dst[(long long)(ty + r) * Q_ + tx] = tile[tx][ty + r];
}

// ---------------------------------------------------------------------------
// Fused kernel: each block owns 64 rows of one batch and the full D=512.
//   Phase 1: wave-per-row softmax (fp32) -> bf16 LDS tile (65 KB).
//   Phase 2: WMMA bf16 GEMM; wave tile 32(M) x 128(N); 16 WMMAs per K-step.
//            B fragments software-pipelined 4 deep in registers; A fragments
//            (LDS) double-buffered one K-step ahead.
// ---------------------------------------------------------------------------
#define AS_STRIDE 520
#define BLK_M 64

__global__ __launch_bounds__(256) void fused_softmax_wmma_gemm(
    const float* __restrict__ sim, const __bf16* __restrict__ ET,
    float* __restrict__ out) {
  __shared__ __attribute__((aligned(16))) __bf16 attnS[BLK_M * AS_STRIDE];

  const int lane = threadIdx.x & 31;
  const int wave = threadIdx.x >> 5;
  const int b    = blockIdx.y;
  const int M0   = blockIdx.x * BLK_M;

  // ---- Phase 1: softmax of 64 rows into LDS (one wave32 per row, 8 passes)
#pragma unroll
  for (int t = 0; t < BLK_M / 8; ++t) {
    const int r = t * 8 + wave;                       // 0..63
    const float* __restrict__ src =
        sim + ((long long)b * C_ + M0 + r) * Q_;

    v4f vals[4];
    float m = -3.402823e38f;
#pragma unroll
    for (int c = 0; c < 4; ++c) {
      vals[c] = __builtin_nontemporal_load(
          (const v4f*)(src + c * 128 + lane * 4));    // streamed once
#pragma unroll
      for (int i = 0; i < 4; ++i) m = fmaxf(m, vals[c][i]);
    }
#pragma unroll
    for (int off = 16; off > 0; off >>= 1)
      m = fmaxf(m, __shfl_xor(m, off, 32));

    float s = 0.0f;
#pragma unroll
    for (int c = 0; c < 4; ++c)
#pragma unroll
      for (int i = 0; i < 4; ++i) {
        float e = __expf(vals[c][i] - m);
        vals[c][i] = e;
        s += e;
      }
#pragma unroll
    for (int off = 16; off > 0; off >>= 1)
      s += __shfl_xor(s, off, 32);
    const float inv = 1.0f / s;

    __bf16* dst = attnS + r * AS_STRIDE;
#pragma unroll
    for (int c = 0; c < 4; ++c) {
      v4bf o;
#pragma unroll
      for (int i = 0; i < 4; ++i) o[i] = (__bf16)(vals[c][i] * inv);
      *(v4bf*)(dst + c * 128 + lane * 4) = o;         // conflict-free ds_store
    }
  }
  __syncthreads();

  // ---- Phase 2: WMMA GEMM. Wave tile = 32(M) x 128(N); 2 M-grp x 4 N-grp.
  const int wm    = wave & 1;           // 0..1  -> M offset wm*32
  const int wn    = wave >> 1;          // 0..3  -> N offset wn*128
  const int halfA = lane >> 4;          // 0/1
  const int nB    = lane & 15;

  const __bf16* __restrict__ arow0 =
      attnS + (wm * 32 + (lane & 15)) * AS_STRIDE + halfA * 16;
  const __bf16* __restrict__ bk =
      ET + (long long)b * D_ * Q_ + (long long)(wn * 128 + nB) * Q_ +
      halfA * 16;

  v8f acc[2][8] = {};
  v16bf bbuf[4];     // circular buffer of B fragments, prefetch distance 4
  v16bf abuf0[2], abuf1[2];

  // Prologue: B fragments f=0..3 of K-step 0, A fragments of K-step 0
#pragma unroll
  for (int j = 0; j < 4; ++j) {
    const __bf16* p = bk + j * (16 * Q_);
    BU u;
    u.h[0] = *(const v8bf*)(p);
    u.h[1] = *(const v8bf*)(p + 8);
    bbuf[j] = u.v;
  }
#pragma unroll
  for (int i = 0; i < 2; ++i) {
    const __bf16* ap = arow0 + i * (16 * AS_STRIDE);
    BU u;
    u.h[0] = *(const v8bf*)(ap);
    u.h[1] = *(const v8bf*)(ap + 8);
    abuf0[i] = u.v;
  }

  // One K-step: consume aUse + bbuf; prefetch A(ks+1) into aPre and B frags
  // 4 ahead. Compile-time buffer indices everywhere (j fully unrolled,
  // outer loop unrolled by 2 for the A parity).
  auto kstep_body = [&](int ks, v16bf (&aUse)[2], v16bf (&aPre)[2]) {
#pragma unroll
    for (int i = 0; i < 2; ++i) {                      // A prefetch (LDS)
      const __bf16* ap = arow0 + i * (16 * AS_STRIDE) + (ks + 1) * 32;
      BU u;
      u.h[0] = *(const v8bf*)(ap);
      u.h[1] = *(const v8bf*)(ap + 8);
      aPre[i] = u.v;
    }
#pragma unroll
    for (int j = 0; j < 8; ++j) {
      const v16bf bv = bbuf[j & 3];
#pragma unroll
      for (int i = 0; i < 2; ++i)
        acc[i][j] = __builtin_amdgcn_wmma_f32_16x16x32_bf16(
            false, aUse[i], false, bv, (short)0, acc[i][j], false, false);
      // prefetch B fragment 4 slots ahead in the (ks, j) stream
      const __bf16* np = (j < 4)
          ? bk + (j + 4) * (16 * Q_) + ks * 32
          : bk + (j - 4) * (16 * Q_) + (ks + 1) * 32;
      BU u;
      u.h[0] = *(const v8bf*)(np);
      u.h[1] = *(const v8bf*)(np + 8);
      bbuf[j & 3] = u.v;
    }
  };

  for (int ks = 0; ks < 14; ks += 2) {
    kstep_body(ks, abuf0, abuf1);
    kstep_body(ks + 1, abuf1, abuf0);
  }
  kstep_body(14, abuf0, abuf1);

  // Epilogue K-step 15: no A prefetch; only the 4 remaining B loads
  {
#pragma unroll
    for (int j = 0; j < 8; ++j) {
      const v16bf bv = bbuf[j & 3];
#pragma unroll
      for (int i = 0; i < 2; ++i)
        acc[i][j] = __builtin_amdgcn_wmma_f32_16x16x32_bf16(
            false, abuf1[i], false, bv, (short)0, acc[i][j], false, false);
      if (j < 4) {                     // frags (15, j+4), consumed at j+4
        const __bf16* np = bk + (j + 4) * (16 * Q_) + 15 * 32;
        BU u;
        u.h[0] = *(const v8bf*)(np);
        u.h[1] = *(const v8bf*)(np + 8);
        bbuf[j & 3] = u.v;
      }
    }
  }

  // ---- Store 32x128 per wave (write-once -> non-temporal, keep L2 for E^T)
#pragma unroll
  for (int i = 0; i < 2; ++i) {
    float* __restrict__ obase =
        out + ((long long)b * C_ + M0 + wm * 32 + i * 16) * D_ + wn * 128;
#pragma unroll
    for (int j = 0; j < 8; ++j)
#pragma unroll
      for (int r = 0; r < 8; ++r)
        __builtin_nontemporal_store(
            acc[i][j][r],
            obase + (long long)(r + halfA * 8) * D_ + j * 16 + nB);
  }
}

// ---------------------------------------------------------------------------
extern "C" void kernel_launch(void* const* d_in, const int* in_sizes, int n_in,
                              void* d_out, int out_size, void* d_ws,
                              size_t ws_size, hipStream_t stream) {
  const float* sim = (const float*)d_in[0];   // [B, C, Q] f32
  const float* enc = (const float*)d_in[1];   // [B, Q, D] f32
  float* out = (float*)d_out;                 // [B, C, D] f32

  // Workspace: E^T bf16 [B, D, Q] (16 MiB)
  __bf16* eT = (__bf16*)d_ws;

  transpose_to_bf16<<<dim3(D_ / 32, Q_ / 32, B_), 256, 0, stream>>>(enc, eT);
  fused_softmax_wmma_gemm<<<dim3(C_ / BLK_M, B_), 256, 0, stream>>>(
      sim, eT, out);
}
